// RNN_52896817218069
// MI455X (gfx1250) — matrix operand order repaired and checked by reading, not compiled, not run on hardware
//
#include <hip/hip_runtime.h>

#define TT 1024
#define BB 64
#define HH 512
#define NWG 4
#define THREADS 256
#define LSTRIDE 552          // halves per LDS row: 512 data + 32 K-ext + 8 pad (bank spread)
#define DT_C 0.5f
#define SQDT_NOISE 0.035355339059327376f   // sqrt(0.5) * 0.05

typedef __attribute__((ext_vector_type(16))) _Float16 v16h;
typedef __attribute__((ext_vector_type(8)))  _Float16 v8h;
typedef __attribute__((ext_vector_type(4)))  _Float16 v4h;
typedef __attribute__((ext_vector_type(8)))  float    v8f;
typedef __attribute__((ext_vector_type(4)))  float    v4f;

__device__ __forceinline__ float ftanh(float x) {
  float ax = __builtin_fabsf(x);
  float e  = __expf(-2.0f * ax);
  float r  = __fdividef(1.0f - e, 1.0f + e);
  return __builtin_copysignf(r, x);
}

__global__ void rnn_init(unsigned* cnt) {
  if (threadIdx.x == 0) *cnt = 0u;
}

__global__ __launch_bounds__(THREADS)
void rnn_persistent(const float* __restrict__ xg,    const float* __restrict__ noiseg,
                    const float* __restrict__ wig,   const float* __restrict__ wrecg,
                    const float* __restrict__ brecg, const float* __restrict__ wog,
                    const float* __restrict__ bog,   const float* __restrict__ h0g,
                    float* __restrict__ outg, unsigned* cnt,
                    float* __restrict__ buf0, float* __restrict__ buf1)
{
  __shared__ _Float16 h_lds[BB * LSTRIDE];      // h_{t-1} (f16) + K-extension columns
  __shared__ float    out_sm[2][16][2];         // double-buffered readout partials

  const int tid   = threadIdx.x;
  const int g     = blockIdx.x;                 // workgroup: owns columns [128g, 128g+128)
  const int lane  = tid & 31;
  const int w     = tid >> 5;                   // wave id 0..7: owns N-tile w
  const int hi    = lane >> 4;                  // half-wave (K/B layout split)
  const int nlane = lane & 15;
  const int jcol  = g * 128 + w * 16 + nlane;   // this lane's output column
  const int wbase = __builtin_amdgcn_readfirstlane(2 * w); // scalar: wave's wo K-tile base

  // ---- Preload stationary B operands into VGPRs (once) -------------------
  v16h breg[17];                                // wrec^T slice + [wi; brec; 0] extension
  #pragma unroll
  for (int kt = 0; kt < 16; ++kt) {
    const v4f* p = (const v4f*)&wrecg[(size_t)jcol * HH + kt * 32 + hi * 16];
    v4f f0 = p[0], f1 = p[1], f2 = p[2], f3 = p[3];
    v16h bv;
    #pragma unroll
    for (int q = 0; q < 4; ++q) {
      bv[q]      = (_Float16)f0[q];
      bv[4 + q]  = (_Float16)f1[q];
      bv[8 + q]  = (_Float16)f2[q];
      bv[12 + q] = (_Float16)f3[q];
    }
    breg[kt] = bv;
  }
  {
    v16h bv;
    #pragma unroll
    for (int q = 0; q < 16; ++q) {
      int k = 512 + hi * 16 + q;
      float val = 0.0f;
      if (k < 515)        val = wig[(k - 512) * HH + jcol];
      else if (k == 515)  val = brecg[jcol];
      bv[q] = (_Float16)val;
    }
    breg[16] = bv;
  }
  v16h bwo[3];                                  // this wave's slice of [wo; 0; 0; 0; bo]
  {
    const int wn = (w == 7) ? 3 : 2;
    #pragma unroll
    for (int q3 = 0; q3 < 3; ++q3) {
      v16h bv;
      #pragma unroll
      for (int q = 0; q < 16; ++q) {
        int k = (2 * w + q3) * 32 + hi * 16 + q;
        float val = 0.0f;
        if (q3 < wn && nlane < 2) {
          if (k < 512)       val = wog[k * 2 + nlane];
          else if (k == 515) val = bog[nlane];
        }
        bv[q] = (_Float16)val;
      }
      bwo[q3] = bv;
    }
  }

  auto ldA = [&](int mt, int kt) -> v16h {      // A fragment from LDS (2x ds_load_b128)
    const int base = (mt * 16 + nlane) * LSTRIDE + kt * 32 + hi * 16;
    v8h lo = *(const v8h*)&h_lds[base];
    v8h hh = *(const v8h*)&h_lds[base + 8];
    v16h a;
    #pragma unroll
    for (int i = 0; i < 8; ++i) { a[i] = lo[i]; a[i + 8] = hh[i]; }
    return a;
  };

  // ---- Persistent time loop ---------------------------------------------
  #pragma clang loop unroll(disable)
  for (int t = 0; t <= TT; ++t) {
    const float* hprev = (t & 1) ? buf0 : buf1;   // holds h_{t-1} (f32)
    float*       hcur  = (t & 1) ? buf1 : buf0;

    // Stage h_{t-1} (or h0 broadcast) into LDS as f16, plus x_t into ext cols.
    #pragma unroll 4
    for (int i = 0; i < 32; ++i) {
      int c4  = (tid + i * THREADS) * 4;          // 8192 chunks of 4 floats
      int row = c4 >> 9, col = c4 & 511;
      v4f f = (t == 0) ? *(const v4f*)&h0g[col] : *(const v4f*)&hprev[c4];
      v4h hv = { (_Float16)f[0], (_Float16)f[1], (_Float16)f[2], (_Float16)f[3] };
      *(v4h*)&h_lds[row * LSTRIDE + col] = hv;
    }
    if (t < TT && tid < BB) {
      const float* xp = &xg[((size_t)tid * TT + t) * 3];
      h_lds[tid * LSTRIDE + 512] = (_Float16)xp[0];
      h_lds[tid * LSTRIDE + 513] = (_Float16)xp[1];
      h_lds[tid * LSTRIDE + 514] = (_Float16)xp[2];
    }
    if (t == 0 && tid < BB) {
      h_lds[tid * LSTRIDE + 515] = (_Float16)1.0f;     // ones column -> brec / bo
      #pragma unroll
      for (int c = 516; c < 544; ++c) h_lds[tid * LSTRIDE + c] = (_Float16)0.0f;
      ((float*)out_sm)[tid] = 0.0f;
    }
    __syncthreads();

    // Prefetch next step's HBM-resident operands (noise slice + x_t+1) into
    // L2 so their miss latency is off the 1024-step serial chain.
    if (t + 1 < TT) {
      int prow = tid & 63;                        // batch row
      int pseg = tid >> 6;                        // 128B segment within 512B slice
      __builtin_prefetch(&noiseg[((size_t)prow * TT + (t + 1)) * HH + g * 128 + pseg * 32], 0, 3);
      if (tid < BB)
        __builtin_prefetch(&xg[((size_t)tid * TT + (t + 1)) * 3], 0, 3);
    }

    // ---- WMMA phase: recurrent GEMM (t < TT) ----------------------------
    v8f acc[4] = { v8f{}, v8f{}, v8f{}, v8f{} };
    if (t < TT) {
      #pragma unroll
      for (int kt = 0; kt <= 16; ++kt) {
        v16h a0 = ldA(0, kt), a1 = ldA(1, kt), a2 = ldA(2, kt), a3 = ldA(3, kt);
        acc[0] = __builtin_amdgcn_wmma_f32_16x16x32_f16(false, a0, false, breg[kt], (short)0, acc[0], false, false);
        acc[1] = __builtin_amdgcn_wmma_f32_16x16x32_f16(false, a1, false, breg[kt], (short)0, acc[1], false, false);
        acc[2] = __builtin_amdgcn_wmma_f32_16x16x32_f16(false, a2, false, breg[kt], (short)0, acc[2], false, false);
        acc[3] = __builtin_amdgcn_wmma_f32_16x16x32_f16(false, a3, false, breg[kt], (short)0, acc[3], false, false);
      }
    }

    // ---- Readout tile: out_{t-1} = h_{t-1} @ [wo; bo] (t > 0) -----------
    // Rides on the same LDS A-fragments; each wave owns <=3 K-tiles, the
    // spare tile's B operand is zero so overlap contributes nothing.
    if (t > 0) {
      v8f accw = {};
      #pragma unroll
      for (int q3 = 0; q3 < 3; ++q3) {
        v16h a = ldA(g, wbase + q3);
        accw = __builtin_amdgcn_wmma_f32_16x16x32_f16(false, a, false, bwo[q3], (short)0, accw, false, false);
      }
      if (nlane < 2) {
        #pragma unroll
        for (int v = 0; v < 8; ++v)
          atomicAdd(&out_sm[t & 1][v + 8 * hi][nlane], accw[v]);
      }
    }

    // ---- Elementwise state update, h_t -> L2 exchange buffer (f32) ------
    if (t < TT) {
      float h0v = (t == 0) ? h0g[jcol] : 0.0f;
      #pragma unroll
      for (int mt = 0; mt < 4; ++mt) {
        #pragma unroll
        for (int v = 0; v < 8; ++v) {
          int bb = mt * 16 + v + 8 * hi;
          float hold = (t == 0) ? h0v : hprev[(size_t)bb * HH + jcol];
          float nz   = noiseg[((size_t)bb * TT + t) * HH + jcol];
          float hnew = (1.0f - DT_C) * hold + DT_C * ftanh(acc[mt][v]) + SQDT_NOISE * nz;
          hcur[(size_t)bb * HH + jcol] = hnew;
        }
      }
    }

    // ---- One device-scope barrier per step ------------------------------
    if (t < TT) {
      __builtin_amdgcn_fence(__ATOMIC_RELEASE, "agent");
      __syncthreads();
      if (tid == 0) {
        __hip_atomic_fetch_add(cnt, 1u, __ATOMIC_RELAXED, __HIP_MEMORY_SCOPE_AGENT);
        unsigned tgt = (unsigned)(NWG * (t + 1));
        while (__hip_atomic_load(cnt, __ATOMIC_RELAXED, __HIP_MEMORY_SCOPE_AGENT) < tgt)
          __builtin_amdgcn_s_sleep(2);
      }
      __syncthreads();
      __builtin_amdgcn_fence(__ATOMIC_ACQUIRE, "agent");   // invalidate WGP$ for fresh h
    } else {
      __syncthreads();
    }

    // emit out[:, t-1, :] (this WG's 16 batch rows), recycle partial buffer
    if (t > 0 && tid < 32) {
      int bl = tid >> 1, o = tid & 1;
      int b  = g * 16 + bl;
      outg[((size_t)b * TT + (t - 1)) * 2 + o] = out_sm[t & 1][bl][o];
      out_sm[t & 1][bl][o] = 0.0f;
    }
  }
}

extern "C" void kernel_launch(void* const* d_in, const int* in_sizes, int n_in,
                              void* d_out, int out_size, void* d_ws, size_t ws_size,
                              hipStream_t stream) {
  (void)in_sizes; (void)n_in; (void)out_size; (void)ws_size;
  const float* x     = (const float*)d_in[0];
  const float* noise = (const float*)d_in[1];
  const float* wi    = (const float*)d_in[2];
  const float* wrec  = (const float*)d_in[3];
  const float* brec  = (const float*)d_in[4];
  const float* wo    = (const float*)d_in[5];
  const float* bo    = (const float*)d_in[6];
  const float* h0    = (const float*)d_in[7];
  float* out = (float*)d_out;

  unsigned* cnt = (unsigned*)d_ws;
  float* buf0 = (float*)((char*)d_ws + 256);
  float* buf1 = buf0 + BB * HH;

  rnn_init<<<1, 64, 0, stream>>>(cnt);
  rnn_persistent<<<NWG, THREADS, 0, stream>>>(x, noise, wi, wrec, brec, wo, bo, h0,
                                              out, cnt, buf0, buf1);
}
